// GATNet_31284541784104
// MI455X (gfx1250) — matrix operand order repaired and checked by reading, not compiled, not run on hardware
//
#include <hip/hip_runtime.h>
#include <hip/hip_bf16.h>

typedef __attribute__((ext_vector_type(16))) __bf16 v16bf;
typedef __attribute__((ext_vector_type(8)))  __bf16 v8bf;
typedef __attribute__((ext_vector_type(8)))  float  v8f;

#define GN 100000
#define GE 1600000
#define GFE 16
#define GH 4
#define GHC 128
#define GEN (GE + GN)

__device__ __forceinline__ unsigned ford(float f) {
    unsigned u = __float_as_uint(f);
    return (u & 0x80000000u) ? ~u : (u | 0x80000000u);
}
__device__ __forceinline__ float frev(unsigned u) {
    return (u & 0x80000000u) ? __uint_as_float(u ^ 0x80000000u) : __uint_as_float(~u);
}

__global__ void k_zero(float* __restrict__ p, long n) {
    long i = (long)blockIdx.x * blockDim.x + threadIdx.x;
    if (i < n) p[i] = 0.0f;
}

// x (f32, row-major [N,128]) -> xbf (bf16, same layout)
__global__ void k_cvt_x(const float* __restrict__ x, __bf16* __restrict__ xbf) {
    long i = (long)blockIdx.x * blockDim.x + threadIdx.x;
    if (i < (long)GN * GHC) xbf[i] = (__bf16)x[i];
}

// W (f32, [K=128][N=128]) -> WT (bf16, transposed [N=128][K=128]) for both Wl, Wr
__global__ void k_cvt_w(const float* __restrict__ Wl, const float* __restrict__ Wr,
                        __bf16* __restrict__ WlT, __bf16* __restrict__ WrT) {
    int i = blockIdx.x * blockDim.x + threadIdx.x;      // over 128*128
    if (i >= GHC * GHC) return;
    int c = i >> 7;          // output row  = column of W
    int k = i & 127;         // output col  = row of W
    WlT[i] = (__bf16)Wl[k * GHC + c];
    WrT[i] = (__bf16)Wr[k * GHC + c];
}

// accumulate incoming edge_attr sums + counts per destination (for self-loop fill)
__global__ void k_edge_mean_accum(const int* __restrict__ dst,
                                  const float* __restrict__ edge_attr,
                                  float* __restrict__ ea_sum,
                                  float* __restrict__ cnt) {
    long idx = (long)blockIdx.x * blockDim.x + threadIdx.x;
    if (idx >= (long)GE * GFE) return;
    int e = (int)(idx >> 4);
    int f = (int)(idx & 15);
    int d = dst[e];
    atomicAdd(ea_sum + (long)d * GFE + f, edge_attr[idx]);
    if (f == 0) atomicAdd(cnt + d, 1.0f);
}

// loop_attr = ea_sum / max(cnt,1)  (in place)
__global__ void k_loop_attr(float* __restrict__ ea_sum, const float* __restrict__ cnt) {
    long idx = (long)blockIdx.x * blockDim.x + threadIdx.x;
    if (idx >= (long)GN * GFE) return;
    float c = cnt[idx >> 4];
    ea_sum[idx] = ea_sum[idx] / fmaxf(c, 1.0f);
}

// xl = x@Wl (grid.y=0) / xr = x@Wr (grid.y=1) via v_wmma_f32_16x16x32_bf16.
// All operands pre-converted to bf16; B pre-transposed so every fragment load is b128.
// Block = 256 threads = 8 waves; wave w computes column-tile w of one 16-row tile.
__global__ void k_gemm_wmma(const __bf16* __restrict__ xbf,
                            const __bf16* __restrict__ WlT,
                            const __bf16* __restrict__ WrT,
                            float* __restrict__ xl,
                            float* __restrict__ xr) {
    const __bf16* __restrict__ WT  = blockIdx.y ? WrT : WlT;
    float* __restrict__        out = blockIdx.y ? xr : xl;

    const int wave  = threadIdx.x >> 5;
    const int lane  = threadIdx.x & 31;
    const int mTile = blockIdx.x;        // 0..6249 (100000/16)
    const int nTile = wave;              // 0..7    (128/16)
    const int lo16  = lane & 15;
    const int hi    = lane >> 4;         // 0/1

    const int row = mTile * 16 + lo16;   // A row for this lane
    const int col = nTile * 16 + lo16;   // B/C/D column for this lane

    const __bf16* __restrict__ arow = xbf + (long)row * GHC;
    const __bf16* __restrict__ bcol = WT  + (long)col * GHC;

    v8f acc = {};
#pragma unroll
    for (int k0 = 0; k0 < 128; k0 += 32) {
        // A fragment: lane(0-15) row M=lane, K=k0+0..7 & k0+16..23;
        //             lane(16-31) same rows, K=k0+8..15 & k0+24..31
        const int ka = k0 + hi * 8;
        v8bf alo = *(const v8bf*)(arow + ka);
        v8bf ahi = *(const v8bf*)(arow + ka + 16);
        v16bf a = __builtin_shufflevector(alo, ahi,
            0, 1, 2, 3, 4, 5, 6, 7, 8, 9, 10, 11, 12, 13, 14, 15);

        // B fragment: lanes 0-15 hold K=k0+0..7 & k0+16..23, lanes 16-31 K=k0+8..15 & k0+24..31; col = lane&15
        const int kb = k0 + hi * 8;
        v8bf blo = *(const v8bf*)(bcol + kb);
        v8bf bhi = *(const v8bf*)(bcol + kb + 16);
        v16bf b = __builtin_shufflevector(blo, bhi,
            0, 1, 2, 3, 4, 5, 6, 7, 8, 9, 10, 11, 12, 13, 14, 15);

        acc = __builtin_amdgcn_wmma_f32_16x16x32_bf16(
            false, a, false, b, (short)0, acc, false, false);
    }

    // D layout: lane holds column `col`, VGPR r -> row mbase+r (hi half rows +8)
    const int mbase = mTile * 16 + hi * 8;
#pragma unroll
    for (int r = 0; r < 8; ++r)
        out[(long)(mbase + r) * GHC + col] = acc[r];
}

// One wave per row of ea2 (edges then self-loops). Fuses ee = ea@We, leaky-relu,
// attention dot, 8-lane head reduction, and segment-max via encoded atomicMax.
__global__ void k_logit(const float* __restrict__ xl, const float* __restrict__ xr,
                        const int* __restrict__ src, const int* __restrict__ dst,
                        const float* __restrict__ edge_attr,
                        const float* __restrict__ loop_attr,
                        const float* __restrict__ We, const float* __restrict__ att,
                        float* __restrict__ logit, unsigned* __restrict__ maxlog) {
    const int wave = threadIdx.x >> 5;
    const int lane = threadIdx.x & 31;
    long e = (long)blockIdx.x * (blockDim.x >> 5) + wave;
    if (e >= GEN) return;

    int s, d;
    const float* __restrict__ ea;
    if (e < GE) { s = src[e]; d = dst[e]; ea = edge_attr + e * GFE; }
    else        { s = d = (int)(e - GE); ea = loop_attr + (long)s * GFE; }

    float eav[GFE];
#pragma unroll
    for (int f = 0; f < GFE; ++f) eav[f] = ea[f];

    const int c0 = lane * 4;                 // 4 channels per lane
    float part = 0.0f;
#pragma unroll
    for (int u = 0; u < 4; ++u) {
        const int c = c0 + u;
        float ee = 0.0f;
#pragma unroll
        for (int f = 0; f < GFE; ++f) ee = fmaf(eav[f], We[f * GHC + c], ee);
        float v = xl[(long)s * GHC + c] + xr[(long)d * GHC + c] + ee;
        v = v > 0.0f ? v : 0.2f * v;         // leaky_relu
        part = fmaf(v, att[c], part);        // att flat [H*C]: index == channel
    }
    // reduce over the 8 lanes of each head (channels h*32..h*32+31 -> lanes h*8..h*8+7)
    part += __shfl_xor(part, 1);
    part += __shfl_xor(part, 2);
    part += __shfl_xor(part, 4);

    if ((lane & 7) == 0) {
        const int h = lane >> 3;
        logit[e * GH + h] = part;
        atomicMax(maxlog + (long)d * GH + h, ford(part));
    }
}

// ev = exp(logit - max[dst]); denom[dst] += ev; store ev in place of logit
__global__ void k_expsum(const int* __restrict__ dst,
                         const unsigned* __restrict__ maxlog,
                         float* __restrict__ alpha,   // holds logits -> ev
                         float* __restrict__ denom) {
    long idx = (long)blockIdx.x * blockDim.x + threadIdx.x;
    if (idx >= (long)GEN * GH) return;
    long e = idx >> 2;
    int  h = (int)(idx & 3);
    int  d = (e < GE) ? dst[e] : (int)(e - GE);
    float m  = frev(maxlog[(long)d * GH + h]);
    float ev = __expf(alpha[idx] - m);
    alpha[idx] = ev;
    atomicAdd(denom + (long)d * GH + h, ev);
}

// alpha = ev/denom[dst]; out[dst] += alpha * xl[src]  (one wave per edge row)
__global__ void k_scatter(const float* __restrict__ xl,
                          const int* __restrict__ src, const int* __restrict__ dst,
                          float* __restrict__ alpha,
                          const float* __restrict__ denom,
                          float* __restrict__ out) {
    const int wave = threadIdx.x >> 5;
    const int lane = threadIdx.x & 31;
    long e = (long)blockIdx.x * (blockDim.x >> 5) + wave;
    if (e >= GEN) return;

    int s, d;
    if (e < GE) { s = src[e]; d = dst[e]; }
    else        { s = d = (int)(e - GE); }

    const int h  = lane >> 3;
    float ev = alpha[e * GH + h];
    float al = ev / denom[(long)d * GH + h];
    if ((lane & 7) == 0) alpha[e * GH + h] = al;   // final normalized alpha

    const int c0 = lane * 4;
#pragma unroll
    for (int u = 0; u < 4; ++u) {
        const int c = c0 + u;
        atomicAdd(out + (long)d * GHC + c, al * xl[(long)s * GHC + c]);
    }
}

__global__ void k_finalize(float* __restrict__ out, const float* __restrict__ bias) {
    long idx = (long)blockIdx.x * blockDim.x + threadIdx.x;
    if (idx >= (long)GN * GHC) return;
    out[idx] = fmaxf(out[idx] + bias[idx & (GHC - 1)], 0.0f);
}

static inline size_t alignup(size_t v) { return (v + 255) & ~(size_t)255; }

extern "C" void kernel_launch(void* const* d_in, const int* in_sizes, int n_in,
                              void* d_out, int out_size, void* d_ws, size_t ws_size,
                              hipStream_t stream) {
    (void)in_sizes; (void)n_in; (void)out_size; (void)ws_size;

    const float* x         = (const float*)d_in[0];
    const int*   ei        = (const int*)  d_in[1];
    const float* edge_attr = (const float*)d_in[2];
    const float* Wl        = (const float*)d_in[3];
    const float* Wr        = (const float*)d_in[4];
    const float* We        = (const float*)d_in[5];
    const float* att       = (const float*)d_in[6];
    const float* bias      = (const float*)d_in[7];
    const int* src = ei;
    const int* dst = ei + GE;

    float* out   = (float*)d_out;                 // [N, H*C]
    float* alpha = out + (size_t)GN * GHC;        // [E+N, H]  (logit -> ev -> alpha)

    char* w = (char*)d_ws;
    float* xl = (float*)w;           w += alignup((size_t)GN * GHC * 4);
    float* xr = (float*)w;           w += alignup((size_t)GN * GHC * 4);
    __bf16* xbf = (__bf16*)w;        w += alignup((size_t)GN * GHC * 2);
    __bf16* WlT = (__bf16*)w;        w += alignup((size_t)GHC * GHC * 2);
    __bf16* WrT = (__bf16*)w;        w += alignup((size_t)GHC * GHC * 2);
    float* ea_sum = (float*)w;       w += alignup((size_t)GN * GFE * 4);  // -> loop_attr
    float* cnt = (float*)w;          w += alignup((size_t)GN * 4);
    float* denom = (float*)w;        w += alignup((size_t)GN * GH * 4);
    unsigned* maxlog = (unsigned*)w; w += alignup((size_t)GN * GH * 4);

    const int BT = 256;
    auto blk = [](long n, int t) { return (unsigned)((n + t - 1) / t); };

    // --- init ---
    k_zero<<<blk((long)GN * GHC, BT), BT, 0, stream>>>(out,   (long)GN * GHC);
    k_zero<<<blk((long)GN * GFE, BT), BT, 0, stream>>>(ea_sum,(long)GN * GFE);
    k_zero<<<blk(GN, BT),             BT, 0, stream>>>(cnt,   GN);
    k_zero<<<blk((long)GN * GH, BT),  BT, 0, stream>>>(denom, (long)GN * GH);
    k_zero<<<blk((long)GN * GH, BT),  BT, 0, stream>>>((float*)maxlog, (long)GN * GH);

    // --- one-time bf16 staging: x, and transposed Wl/Wr for contiguous B fragments ---
    k_cvt_x<<<blk((long)GN * GHC, BT), BT, 0, stream>>>(x, xbf);
    k_cvt_w<<<blk(GHC * GHC, BT), BT, 0, stream>>>(Wl, Wr, WlT, WrT);

    // --- self-loop attr = mean of incoming edge_attr ---
    k_edge_mean_accum<<<blk((long)GE * GFE, BT), BT, 0, stream>>>(dst, edge_attr, ea_sum, cnt);
    k_loop_attr<<<blk((long)GN * GFE, BT), BT, 0, stream>>>(ea_sum, cnt);

    // --- xl = x@Wl, xr = x@Wr via WMMA (6250 m-tiles, 8 waves = 8 n-tiles each) ---
    dim3 ggrid(GN / 16, 2);
    k_gemm_wmma<<<ggrid, 256, 0, stream>>>(xbf, WlT, WrT, xl, xr);

    // --- attention logits + segment max (wave per edge row) ---
    unsigned ewaves = blk(GEN, 8);
    k_logit<<<ewaves, 256, 0, stream>>>(xl, xr, src, dst, edge_attr, ea_sum,
                                        We, att, alpha, maxlog);

    // --- softmax exp + segment sum ---
    k_expsum<<<blk((long)GEN * GH, BT), BT, 0, stream>>>(dst, maxlog, alpha, denom);

    // --- normalize + weighted scatter-aggregate ---
    k_scatter<<<ewaves, 256, 0, stream>>>(xl, src, dst, alpha, denom, out);

    // --- bias + relu ---
    k_finalize<<<blk((long)GN * GHC, BT), BT, 0, stream>>>(out, bias);
}
